// Net_71940702208553
// MI455X (gfx1250) — matrix operand order repaired and checked by reading, not compiled, not run on hardware
//
#include <hip/hip_runtime.h>
#include <hip/hip_bf16.h>

#define N_NODES 20000
#define N_EDGES 320000
#define F_IN    128
#define C_DIM   300
#define CP      320   // padded channel dim (multiple of 64)
#define GP      960   // padded 3C (multiple of 64)
#define L_STEPS 8

// LDS B-panel row stride (bf16 elements): 328 elems = 656 B = 164 dwords.
// 164 mod 64 = 36, gcd(36,64)=4 -> 16 column-lanes hit 16 distinct 4-bank groups.
#define SB_STRIDE 328

typedef __attribute__((ext_vector_type(16))) __bf16 v16bf;
typedef __attribute__((ext_vector_type(8)))  __bf16 v8bf;
typedef __attribute__((ext_vector_type(8)))  float  v8f;

// ---------------------------------------------------------------- utilities
__global__ void k_fill_zero(float* __restrict__ p, long n) {
  long i = (long)blockIdx.x * blockDim.x + threadIdx.x;
  if (i < n) p[i] = 0.f;
}

__global__ void k_pad_x(const float* __restrict__ x, float* __restrict__ H) {
  long i = (long)blockIdx.x * blockDim.x + threadIdx.x;   // over N*CP
  if (i >= (long)N_NODES * CP) return;
  int n = (int)(i / CP), c = (int)(i % CP);
  H[i] = (c < F_IN) ? x[(long)n * F_IN + c] : 0.f;
}

__global__ void k_relu(float* __restrict__ H, long n) {
  long i = (long)blockIdx.x * blockDim.x + threadIdx.x;
  if (i < n) H[i] = fmaxf(H[i], 0.f);
}

// W [L,300,300] fp32 -> Wt [L,320,320] bf16, transposed: Wt[l][j][k] = W[l][k][j]
__global__ void k_prep_W(const float* __restrict__ W, __bf16* __restrict__ Wt, int Lcnt) {
  long i = (long)blockIdx.x * blockDim.x + threadIdx.x;   // over Lcnt*CP*CP
  if (i >= (long)Lcnt * CP * CP) return;
  int l = (int)(i / (CP * CP));
  int r = (int)(i % (CP * CP));
  int j = r / CP, k = r % CP;
  float v = (j < C_DIM && k < C_DIM)
              ? W[(long)l * C_DIM * C_DIM + (long)k * C_DIM + j] : 0.f;
  Wt[i] = (__bf16)v;
}

// Wg [900,300] fp32 -> Bt [960,320] bf16: Bt[g*320+c][k] = Wg[g*300+c][k]
__global__ void k_prep_gru(const float* __restrict__ Wg, __bf16* __restrict__ Wt) {
  long i = (long)blockIdx.x * blockDim.x + threadIdx.x;   // over GP*CP
  if (i >= (long)GP * CP) return;
  int j = (int)(i / CP), k = (int)(i % CP);
  int g = j / CP, c = j % CP;
  float v = (c < C_DIM && k < C_DIM)
              ? Wg[((long)(g * C_DIM + c)) * C_DIM + k] : 0.f;
  Wt[i] = (__bf16)v;
}

// ---------------------------------------------------------------- WMMA GEMM
// C[M x Npad] = A[M x 320] (fp32, cvt->bf16) * B[320 x Npad]
// B supplied transposed: Bt[j][k] (bf16, row-major, ld = CP).
// Block = 8 waves -> 128 rows x 64 cols. The 64x320 bf16 B panel (contiguous
// 40960 B in the transposed layout) is staged in LDS once per block via the
// CDNA5 async global->LDS path, then all 8 waves feed WMMAs from LDS.
__global__ void __launch_bounds__(256)
k_gemm_wmma(const float* __restrict__ A, const __bf16* __restrict__ Bt,
            float* __restrict__ Cout, int colGroups, int ldc) {
  __shared__ __bf16 sB[64 * SB_STRIDE];          // 41984 B

  const int tid  = threadIdx.x;
  const int wave = tid >> 5;
  const int lane = tid & 31;
  const int rg = blockIdx.x / colGroups;          // row group (8 row tiles)
  const int cg = blockIdx.x - rg * colGroups;
  const int colBase = cg * 64;

  // ---- async copy of B panel into LDS (2560 x 16B chunks, 10 per thread)
  {
    const char* gpanel = (const char*)(Bt + (size_t)colBase * CP);
    unsigned sbase = (unsigned)(unsigned long long)(uintptr_t)&sB[0];
#pragma unroll
    for (int it = 0; it < 10; ++it) {
      int c = tid + it * 256;                     // 0..2559, no divergence
      int j = c / 40;                             // panel row
      int o = c - j * 40;                         // 16B chunk within row
      unsigned ldst = sbase + (unsigned)(j * (SB_STRIDE * 2) + o * 16);
      unsigned long long gsrc = (unsigned long long)(uintptr_t)(gpanel + c * 16);
      asm volatile("global_load_async_to_lds_b128 %0, %1, off"
                   :: "v"(ldst), "v"(gsrc) : "memory");
    }
    asm volatile("s_wait_asynccnt 0x0" ::: "memory");
  }
  __syncthreads();

  const int rowTiles = N_NODES / 16;              // 1250 (exact)
  const int rowTile = rg * 8 + wave;
  if (rowTile >= rowTiles) return;                // wave-uniform; after barrier

  const int rowBase = rowTile * 16;
  const int r  = lane & 15;
  const int hi = lane >> 4;

  const float* arow = A + (long)(rowBase + r) * CP;
  v8f acc0 = {}, acc1 = {}, acc2 = {}, acc3 = {};

  for (int kb = 0; kb < CP; kb += 32) {
    // A fragment (16x32 bf16): lanes 0-15 rows M=r get K = kb+{0..15},
    // lanes 16-31 get K = kb+{16..31}  -> two contiguous 8-float chunks
    const float* ap0 = arow + kb + 16 * hi;
    const float* ap1 = arow + kb + 16 * hi + 8;
    v16bf a;
#pragma unroll
    for (int e = 0; e < 8; ++e) {
      a[e]     = (__bf16)ap0[e];
      a[e + 8] = (__bf16)ap1[e];
    }
    // B fragments (32x16 bf16) from LDS: lane covers col n=r of each 16-col
    // tile; K = kb + 16*hi + e, contiguous in the panel row -> 2x 16B ds loads
    const __bf16* bb = &sB[r * SB_STRIDE + kb + 16 * hi];
    union { v16bf v; v8bf h[2]; } u0, u1, u2, u3;
    u0.h[0] = *(const v8bf*)(bb);
    u0.h[1] = *(const v8bf*)(bb + 8);
    u1.h[0] = *(const v8bf*)(bb + 16 * SB_STRIDE);
    u1.h[1] = *(const v8bf*)(bb + 16 * SB_STRIDE + 8);
    u2.h[0] = *(const v8bf*)(bb + 32 * SB_STRIDE);
    u2.h[1] = *(const v8bf*)(bb + 32 * SB_STRIDE + 8);
    u3.h[0] = *(const v8bf*)(bb + 48 * SB_STRIDE);
    u3.h[1] = *(const v8bf*)(bb + 48 * SB_STRIDE + 8);

    acc0 = __builtin_amdgcn_wmma_f32_16x16x32_bf16(false, a, false, u0.v, (short)0, acc0, false, false);
    acc1 = __builtin_amdgcn_wmma_f32_16x16x32_bf16(false, a, false, u1.v, (short)0, acc1, false, false);
    acc2 = __builtin_amdgcn_wmma_f32_16x16x32_bf16(false, a, false, u2.v, (short)0, acc2, false, false);
    acc3 = __builtin_amdgcn_wmma_f32_16x16x32_bf16(false, a, false, u3.v, (short)0, acc3, false, false);
  }

  // C/D layout: lane (n=r, hi), VGPR v -> C[M = 8*hi + v][n]
  float* crow = Cout + (long)(rowBase + 8 * hi) * ldc + colBase + r;
#pragma unroll
  for (int v = 0; v < 8; ++v) {
    crow[(long)v * ldc +  0] = acc0[v];
    crow[(long)v * ldc + 16] = acc1[v];
    crow[(long)v * ldc + 32] = acc2[v];
    crow[(long)v * ldc + 48] = acc3[v];
  }
}

// ---------------------------------------------------------------- scatter add
// AGG[dst[e]] += M[src[e]] over all edges; 4 features per thread.
__global__ void k_scatter(const float* __restrict__ Mb, const int* __restrict__ src,
                          const int* __restrict__ dst, float* __restrict__ AGG) {
  long i = (long)blockIdx.x * blockDim.x + threadIdx.x;   // over N_EDGES * (CP/4)
  if (i >= (long)N_EDGES * (CP / 4)) return;
  int e = (int)(i / (CP / 4));
  int f = (int)(i % (CP / 4)) * 4;
  int s = src[e], d = dst[e];
  const float4 v = *(const float4*)(Mb + (long)s * CP + f);
  float* out = AGG + (long)d * CP + f;
  atomicAdd(out + 0, v.x);
  atomicAdd(out + 1, v.y);
  atomicAdd(out + 2, v.z);
  atomicAdd(out + 3, v.w);
}

// ---------------------------------------------------------------- GRU cell
__global__ void k_gru(const float* __restrict__ GI, const float* __restrict__ GH,
                      const float* __restrict__ bih, const float* __restrict__ bhh,
                      float* __restrict__ H) {
  long i = (long)blockIdx.x * blockDim.x + threadIdx.x;   // over N*C_DIM
  if (i >= (long)N_NODES * C_DIM) return;
  int n = (int)(i / C_DIM), c = (int)(i % C_DIM);
  const float* gi = GI + (long)n * GP;
  const float* gh = GH + (long)n * GP;
  float ir  = gi[c]          + bih[c];
  float hr  = gh[c]          + bhh[c];
  float iz  = gi[CP + c]     + bih[C_DIM + c];
  float hz  = gh[CP + c]     + bhh[C_DIM + c];
  float inn = gi[2 * CP + c] + bih[2 * C_DIM + c];
  float hn  = gh[2 * CP + c] + bhh[2 * C_DIM + c];
  float rg = 1.f / (1.f + __expf(-(ir + hr)));
  float zg = 1.f / (1.f + __expf(-(iz + hz)));
  float ng = tanhf(inn + rg * hn);
  float* hp = H + (long)n * CP + c;
  float h = *hp;
  *hp = (1.f - zg) * ng + zg * h;
}

// ---------------------------------------------------------------- pooling
__global__ void k_colsum(const float* __restrict__ H, float* __restrict__ pooled) {
  int tid = threadIdx.x;                                  // blockDim = CP (320)
  float s = 0.f;
  for (int n = blockIdx.x; n < N_NODES; n += gridDim.x)
    s += H[(long)n * CP + tid];                           // coalesced rows
  if (tid < C_DIM) atomicAdd(&pooled[tid], s);
}

__global__ void k_logsoftmax(const float* __restrict__ pooled, float* __restrict__ out) {
  __shared__ float red[256];
  int tid = threadIdx.x;                                  // blockDim = 256
  float v0 = (tid < C_DIM)       ? pooled[tid]       * (1.f / N_NODES) : -1e30f;
  float v1 = (tid + 256 < C_DIM) ? pooled[tid + 256] * (1.f / N_NODES) : -1e30f;
  red[tid] = fmaxf(v0, v1);
  __syncthreads();
  for (int s = 128; s > 0; s >>= 1) {
    if (tid < s) red[tid] = fmaxf(red[tid], red[tid + s]);
    __syncthreads();
  }
  float mx = red[0];
  __syncthreads();
  float e0 = (tid < C_DIM)       ? __expf(v0 - mx) : 0.f;
  float e1 = (tid + 256 < C_DIM) ? __expf(v1 - mx) : 0.f;
  red[tid] = e0 + e1;
  __syncthreads();
  for (int s = 128; s > 0; s >>= 1) {
    if (tid < s) red[tid] += red[tid + s];
    __syncthreads();
  }
  float lse = __logf(red[0]);
  if (tid < C_DIM)       out[tid]       = v0 - mx - lse;
  if (tid + 256 < C_DIM) out[tid + 256] = v1 - mx - lse;
}

// ---------------------------------------------------------------- launch
extern "C" void kernel_launch(void* const* d_in, const int* in_sizes, int n_in,
                              void* d_out, int out_size, void* d_ws, size_t ws_size,
                              hipStream_t stream) {
  const float* x    = (const float*)d_in[0];
  const int*   eidx = (const int*)d_in[1];
  const int*   src  = eidx;
  const int*   dst  = eidx + N_EDGES;
  const float* W1   = (const float*)d_in[2];
  const float* wih1 = (const float*)d_in[3];
  const float* whh1 = (const float*)d_in[4];
  const float* bih1 = (const float*)d_in[5];
  const float* bhh1 = (const float*)d_in[6];
  const float* W2   = (const float*)d_in[7];
  const float* wih2 = (const float*)d_in[8];
  const float* whh2 = (const float*)d_in[9];
  const float* bih2 = (const float*)d_in[10];
  const float* bhh2 = (const float*)d_in[11];
  float* out = (float*)d_out;

  char* wsp = (char*)d_ws;
  auto alloc = [&](size_t bytes) -> char* {
    char* p = wsp;
    wsp += (bytes + 255) & ~(size_t)255;
    return p;
  };
  float*  H     = (float*)alloc(sizeof(float) * (size_t)N_NODES * CP);
  float*  Mb    = (float*)alloc(sizeof(float) * (size_t)N_NODES * CP);
  float*  AGG   = (float*)alloc(sizeof(float) * (size_t)N_NODES * CP);
  float*  GI    = (float*)alloc(sizeof(float) * (size_t)N_NODES * GP);
  float*  GH    = (float*)alloc(sizeof(float) * (size_t)N_NODES * GP);
  __bf16* Wt1   = (__bf16*)alloc(sizeof(__bf16) * (size_t)L_STEPS * CP * CP);
  __bf16* Wt2   = (__bf16*)alloc(sizeof(__bf16) * (size_t)L_STEPS * CP * CP);
  __bf16* wih1T = (__bf16*)alloc(sizeof(__bf16) * (size_t)GP * CP);
  __bf16* whh1T = (__bf16*)alloc(sizeof(__bf16) * (size_t)GP * CP);
  __bf16* wih2T = (__bf16*)alloc(sizeof(__bf16) * (size_t)GP * CP);
  __bf16* whh2T = (__bf16*)alloc(sizeof(__bf16) * (size_t)GP * CP);
  float*  pooled = (float*)alloc(sizeof(float) * CP);

  auto nblk = [](long n) { return (int)((n + 255) / 256); };

  // weight prep (bf16, transposed, zero-padded)
  { long n = (long)L_STEPS * CP * CP;
    k_prep_W<<<nblk(n), 256, 0, stream>>>(W1, Wt1, L_STEPS);
    k_prep_W<<<nblk(n), 256, 0, stream>>>(W2, Wt2, L_STEPS); }
  { long n = (long)GP * CP;
    k_prep_gru<<<nblk(n), 256, 0, stream>>>(wih1, wih1T);
    k_prep_gru<<<nblk(n), 256, 0, stream>>>(whh1, whh1T);
    k_prep_gru<<<nblk(n), 256, 0, stream>>>(wih2, wih2T);
    k_prep_gru<<<nblk(n), 256, 0, stream>>>(whh2, whh2T); }

  // h0 = zero-padded x
  { long n = (long)N_NODES * CP;
    k_pad_x<<<nblk(n), 256, 0, stream>>>(x, H); }

  const int rowGroups = (N_NODES / 16 + 7) / 8;           // 157 (8 tiles/block)
  auto gemm = [&](const float* A, const __bf16* Bt, float* Cp, int Npad) {
    int colGroups = Npad / 64;
    k_gemm_wmma<<<rowGroups * colGroups, 256, 0, stream>>>(A, Bt, Cp, colGroups, Npad);
  };

  for (int layer = 0; layer < 2; ++layer) {
    const __bf16* Wt   = layer ? Wt2 : Wt1;
    const __bf16* wihT = layer ? wih2T : wih1T;
    const __bf16* whhT = layer ? whh2T : whh1T;
    const float*  bih  = layer ? bih2 : bih1;
    const float*  bhh  = layer ? bhh2 : bhh1;
    for (int i = 0; i < L_STEPS; ++i) {
      gemm(H, Wt + (size_t)i * CP * CP, Mb, CP);          // m = h @ W[i]
      { long n = (long)N_NODES * CP;
        k_fill_zero<<<nblk(n), 256, 0, stream>>>(AGG, n); }
      { long n = (long)N_EDGES * (CP / 4);
        k_scatter<<<nblk(n), 256, 0, stream>>>(Mb, src, dst, AGG); }
      gemm(AGG, wihT, GI, GP);                            // gi = agg @ w_ih^T
      gemm(H,   whhT, GH, GP);                            // gh = h   @ w_hh^T
      { long n = (long)N_NODES * C_DIM;
        k_gru<<<nblk(n), 256, 0, stream>>>(GI, GH, bih, bhh, H); }
    }
    if (layer == 0) {
      long n = (long)N_NODES * CP;
      k_relu<<<nblk(n), 256, 0, stream>>>(H, n);
    }
  }

  k_fill_zero<<<nblk(CP), 256, 0, stream>>>(pooled, CP);
  k_colsum<<<200, CP, 0, stream>>>(H, pooled);
  k_logsoftmax<<<1, 256, 0, stream>>>(pooled, out);
}
